// LearnableDotProdAttention_68865505624333
// MI455X (gfx1250) — matrix operand order repaired and checked by reading, not compiled, not run on hardware
//
#include <hip/hip_runtime.h>

// ---------------------------------------------------------------------------
// Bilinear attention  out = softmax(mask(Q W K^T)) V    (B=8,Q=K=2048,D=512)
// CDNA5 / gfx1250, wave32. bf16 WMMA (f32 accum) fused flash-attention with
// double-buffered async global->LDS staging.
// ---------------------------------------------------------------------------

typedef __bf16 bf16_t;
typedef __attribute__((ext_vector_type(16))) __bf16 v16bf;
typedef __attribute__((ext_vector_type(8)))  float  v8f;

static_assert(sizeof(v16bf) == 32, "v16bf must be 32 bytes");
static_assert(sizeof(v8f)   == 32, "v8f must be 32 bytes");

constexpr int Bc = 8;
constexpr int Qc = 2048;
constexpr int Kc = 2048;
constexpr int Dc = 512;
constexpr float MASK_VALUE = -1e-06f;   // faithful to reference (NOT -1e6)

#define WMMA_BF16(A, Bm, C) \
  __builtin_amdgcn_wmma_f32_16x16x32_bf16(false, (A), false, (Bm), (short)0, (C), false, false)

// ---- CDNA5 async global->LDS copy (ASYNCcnt) with portable fallback --------
#if defined(__has_builtin)
#if __has_builtin(__builtin_amdgcn_global_load_async_to_lds_b128)
#define HAVE_ASYNC_LDS 1
#endif
#endif

#if defined(HAVE_ASYNC_LDS)
typedef int v4i_t __attribute__((vector_size(4 * sizeof(int))));
typedef __attribute__((address_space(1))) v4i_t* as1_v4i_ptr;
typedef __attribute__((address_space(3))) v4i_t* as3_v4i_ptr;
#endif

__device__ inline void async_copy16(const void* g, void* l) {
#if defined(HAVE_ASYNC_LDS)
  __builtin_amdgcn_global_load_async_to_lds_b128((as1_v4i_ptr)g, (as3_v4i_ptr)l, 0, 0);
#else
  *(uint4*)l = *(const uint4*)g;
#endif
}

__device__ inline void async_wait() {
#if defined(HAVE_ASYNC_LDS)
#if __has_builtin(__builtin_amdgcn_s_wait_asynccnt)
  __builtin_amdgcn_s_wait_asynccnt(0);
#else
  asm volatile("s_wait_asynccnt 0" ::: "memory");
#endif
#endif
}

// Load a 16-bit A-fragment (16x32, contiguous k order, k = 16*half + s): two contiguous 16B chunks.
__device__ inline v16bf ld_frag(const bf16_t* p_lo, const bf16_t* p_hi) {
  union { v16bf v; uint4 q[2]; } u;
  u.q[0] = *(const uint4*)p_lo;
  u.q[1] = *(const uint4*)p_hi;
  return u.v;
}

// ---------------------------------------------------------------------------
// fp32 -> bf16 conversion, 8 elements / thread, 16B packed stores
// ---------------------------------------------------------------------------
__global__ __launch_bounds__(256) void cvt_kernel(const float* __restrict__ src,
                                                  bf16_t* __restrict__ dst, int n8) {
  int i = blockIdx.x * blockDim.x + threadIdx.x;
  if (i >= n8) return;
  const float4* sp = (const float4*)src;
  float4 a = sp[2 * i], b = sp[2 * i + 1];
  union { uint4 q; bf16_t e[8]; } u;
  u.e[0] = (bf16_t)a.x; u.e[1] = (bf16_t)a.y; u.e[2] = (bf16_t)a.z; u.e[3] = (bf16_t)a.w;
  u.e[4] = (bf16_t)b.x; u.e[5] = (bf16_t)b.y; u.e[6] = (bf16_t)b.z; u.e[7] = (bf16_t)b.w;
  ((uint4*)dst)[i] = u.q;
}

// W (fp32 row-major [d][n]) -> Wt (bf16 col-major [n][d])
__global__ __launch_bounds__(256) void wt_kernel(const float* __restrict__ w,
                                                 bf16_t* __restrict__ wt) {
  int i = blockIdx.x * 256 + threadIdx.x;   // 0 .. 512*512-1
  int d = i >> 9, n = i & 511;
  wt[(size_t)n * Dc + d] = (bf16_t)w[i];
}

// ---------------------------------------------------------------------------
// QW = queries @ W  (bf16 staged inputs, WMMA bf16 / f32 accumulate)
// grid: (B*Q/16, 4), block 256 (8 waves). wave -> one 16x16 output tile.
// A/B fragments are contiguous 32B global loads.
// ---------------------------------------------------------------------------
__global__ __launch_bounds__(256) void qw_kernel(const bf16_t* __restrict__ qbf,
                                                 const bf16_t* __restrict__ wt,
                                                 bf16_t* __restrict__ qwbf) {
  const int tid  = threadIdx.x;
  const int lane = tid & 31;
  const int wave = tid >> 5;
  const int half = lane >> 4;
  const int ln   = lane & 15;
  const int row0  = blockIdx.x * 16;                 // flat row over B*Q
  const int dout0 = (blockIdx.y * 8 + wave) * 16;    // output column tile

  v8f c;
#pragma unroll
  for (int r = 0; r < 8; ++r) c[r] = 0.0f;

  const bf16_t* qrow = qbf + (size_t)(row0 + ln) * Dc;
  const bf16_t* wcol = wt + (size_t)(dout0 + ln) * Dc;
#pragma unroll
  for (int dk = 0; dk < Dc; dk += 32) {
    v16bf a  = ld_frag(qrow + dk + 16 * half, qrow + dk + 16 * half + 8);
    v16bf bb = *(const v16bf*)(wcol + dk + 16 * half);   // B[k][n] = Wt[n][k]
    c = WMMA_BF16(a, bb, c);
  }
#pragma unroll
  for (int r = 0; r < 8; ++r)
    qwbf[(size_t)(row0 + r + 8 * half) * Dc + dout0 + ln] = (bf16_t)c[r];
}

// ---------------------------------------------------------------------------
// Fused flash attention, double-buffered staging.
// block = 256 threads = 8 waves; 4 row-groups (16 Q rows) x 2 D-halves (256).
// ---------------------------------------------------------------------------
constexpr size_t SM_QW   = 0;                                       // 65536
constexpr size_t SM_K    = SM_QW + 64 * Dc * sizeof(bf16_t);        // 2x32768
constexpr size_t SM_V    = SM_K  + 2 * 32 * Dc * sizeof(bf16_t);    // 2x32768
constexpr size_t SM_SRED = SM_V  + 2 * 32 * Dc * sizeof(bf16_t);    // 8192
constexpr size_t SM_P    = SM_SRED + 4 * 16 * 32 * sizeof(float);   // 4096
constexpr size_t SM_SCL  = SM_P  + 4 * 16 * 32 * sizeof(bf16_t);    // 256
constexpr size_t SM_TOT  = SM_SCL + 4 * 16 * sizeof(float);         // 209152

// V tile global -> registers (two adjacent rows per thread, 4 chunk passes)
__device__ inline void load_v_regs(const bf16_t* gvb, int tid, uint4 vreg[8]) {
  const uint4* gv = (const uint4*)gvb;
#pragma unroll
  for (int i = 0; i < 4; ++i) {
    int idx = tid + i * 256;         // 0..1023
    int kp  = idx >> 6;              // row pair 0..15
    int d8q = idx & 63;              // d-chunk (8 elems)
    vreg[2 * i]     = gv[(size_t)(2 * kp)     * 64 + d8q];
    vreg[2 * i + 1] = gv[(size_t)(2 * kp + 1) * 64 + d8q];
  }
}
// registers -> LDS transposed [d][k], packed bf16 pairs (4B stores)
__device__ inline void store_v_lds(bf16_t* lsv, int tid, const uint4 vreg[8]) {
  unsigned* lv = (unsigned*)lsv;
#pragma unroll
  for (int i = 0; i < 4; ++i) {
    int idx = tid + i * 256;
    int kp  = idx >> 6;
    int d8  = (idx & 63) << 3;
    union { uint4 q; bf16_t e[8]; } a, b;
    a.q = vreg[2 * i]; b.q = vreg[2 * i + 1];
#pragma unroll
    for (int j = 0; j < 8; ++j) {
      union { unsigned u; bf16_t e[2]; } p;
      p.e[0] = a.e[j]; p.e[1] = b.e[j];
      lv[(d8 + j) * 16 + kp] = p.u;
    }
  }
}

__global__ __launch_bounds__(256) void fa_kernel(const bf16_t* __restrict__ qw,
                                                 const bf16_t* __restrict__ kbf,
                                                 const bf16_t* __restrict__ vbf,
                                                 const int* __restrict__ masks,
                                                 float* __restrict__ out) {
  extern __shared__ char smem[];
  bf16_t* ls_qw   = (bf16_t*)(smem + SM_QW);
  bf16_t* ls_k    = (bf16_t*)(smem + SM_K);     // 2 x [32][512] row-major
  bf16_t* ls_v    = (bf16_t*)(smem + SM_V);     // 2 x transposed [512][32]
  float*  ls_sred = (float*)(smem + SM_SRED);   // [4][2][8][32]
  bf16_t* ls_p    = (bf16_t*)(smem + SM_P);     // [4][16][32]
  float*  ls_scl  = (float*)(smem + SM_SCL);    // [4][16]

  const int tid  = threadIdx.x;
  const int lane = tid & 31;
  const int wave = tid >> 5;
  const int qg   = wave >> 1;   // row-group 0..3
  const int h    = wave & 1;    // D-half 0..1
  const int half = lane >> 4;
  const int ln   = lane & 15;

  const int b  = blockIdx.x >> 5;          // 32 q-tiles per batch
  const int q0 = (blockIdx.x & 31) * 64;
  const int mlen = masks[b];

  // ---- stage QW tile (64x512 bf16) ----
  {
    const uint4* src = (const uint4*)(qw + ((size_t)b * Qc + q0) * Dc);
    uint4* dst = (uint4*)ls_qw;
#pragma unroll
    for (int i = 0; i < 16; ++i) dst[tid + i * 256] = src[tid + i * 256];
  }

  // ---- prologue: tile 0 into buffer 0 (K async, V transposed) ----
  {
    const bf16_t* gk = kbf + (size_t)b * Kc * Dc;
#pragma unroll
    for (int i = 0; i < 8; ++i) {
      size_t off = (size_t)(tid + i * 256) * 8;
      async_copy16(gk + off, ls_k + off);
    }
    uint4 vreg[8];
    load_v_regs(vbf + (size_t)b * Kc * Dc, tid, vreg);
    store_v_lds(ls_v, tid, vreg);
  }
  async_wait();
  __syncthreads();

  // ---- invariant A fragments of QW for this wave's 16 rows x 256 d-slice ----
  v16bf aS[8];
  {
    const bf16_t* qrow = ls_qw + (qg * 16 + ln) * Dc;
#pragma unroll
    for (int c = 0; c < 8; ++c) {
      int dbase = h * 256 + 32 * c;
      aS[c] = ld_frag(qrow + dbase + 16 * half, qrow + dbase + 16 * half + 8);
    }
  }

  v8f o[16];
#pragma unroll
  for (int t = 0; t < 16; ++t)
#pragma unroll
    for (int r = 0; r < 8; ++r) o[t][r] = 0.0f;

  float mrow[8], lrow[8];
#pragma unroll
  for (int r = 0; r < 8; ++r) { mrow[r] = -3.0e38f; lrow[r] = 0.0f; }

  for (int it = 0; it < Kc / 32; ++it) {
    const int kb   = it * 32;
    const int cur  = it & 1;
    const int nxt  = cur ^ 1;
    const bool more = (it + 1) < (Kc / 32);
    const bf16_t* lk = ls_k + cur * (32 * Dc);
    const bf16_t* lv = ls_v + cur * (32 * Dc);

    // fire async K copy for next tile; load next V tile into registers
    if (more) {
      const bf16_t* gk = kbf + ((size_t)b * Kc + kb + 32) * Dc;
      bf16_t* lkn = ls_k + nxt * (32 * Dc);
#pragma unroll
      for (int i = 0; i < 8; ++i) {
        size_t off = (size_t)(tid + i * 256) * 8;
        async_copy16(gk + off, lkn + off);
      }
    }
    uint4 vreg[8];
    if (more) load_v_regs(vbf + ((size_t)b * Kc + kb + 32) * Dc, tid, vreg);

    // ---- partial S over this wave's 256-wide d slice ----
    v8f s0, s1;
#pragma unroll
    for (int r = 0; r < 8; ++r) { s0[r] = 0.0f; s1[r] = 0.0f; }
#pragma unroll
    for (int c = 0; c < 8; ++c) {
      int dbase = h * 256 + 32 * c;
      v16bf b0 = *(const v16bf*)(lk + (ln)*Dc      + dbase + 16 * half);
      v16bf b1 = *(const v16bf*)(lk + (16 + ln)*Dc + dbase + 16 * half);
      s0 = WMMA_BF16(aS[c], b0, s0);
      s1 = WMMA_BF16(aS[c], b1, s1);
    }

    float* sr = ls_sred + qg * 512;
    if (h == 0) {
#pragma unroll
      for (int r = 0; r < 8; ++r) {
        sr[r * 32 + lane]       = s0[r];
        sr[256 + r * 32 + lane] = s1[r];
      }
    }
    __syncthreads();

    if (h == 1) {
      // combine partner's partial sums -> full S(16x32)
#pragma unroll
      for (int r = 0; r < 8; ++r) {
        s0[r] += sr[r * 32 + lane];
        s1[r] += sr[256 + r * 32 + lane];
      }
      // length mask (MASK_VALUE, not -inf: masked cols still enter softmax)
      bool v0 = (kb + ln) < mlen;
      bool v1 = (kb + 16 + ln) < mlen;
#pragma unroll
      for (int r = 0; r < 8; ++r) {
        if (!v0) s0[r] = MASK_VALUE;
        if (!v1) s1[r] = MASK_VALUE;
      }
      // online softmax; xor<=8 shuffles stay within each 16-lane C-tile half
      float scl[8];
#pragma unroll
      for (int r = 0; r < 8; ++r) {
        float x = fmaxf(s0[r], s1[r]);
        x = fmaxf(x, __shfl_xor(x, 1));
        x = fmaxf(x, __shfl_xor(x, 2));
        x = fmaxf(x, __shfl_xor(x, 4));
        x = fmaxf(x, __shfl_xor(x, 8));
        float mnew = fmaxf(mrow[r], x);
        scl[r]  = __expf(mrow[r] - mnew);
        mrow[r] = mnew;
        float p0 = __expf(s0[r] - mnew);
        float p1 = __expf(s1[r] - mnew);
        s0[r] = p0; s1[r] = p1;
        float rs = p0 + p1;
        rs += __shfl_xor(rs, 1);
        rs += __shfl_xor(rs, 2);
        rs += __shfl_xor(rs, 4);
        rs += __shfl_xor(rs, 8);
        lrow[r] = lrow[r] * scl[r] + rs;
      }
      bf16_t* pp = ls_p + qg * 512;
#pragma unroll
      for (int r = 0; r < 8; ++r) {
        int row = r + 8 * half;
        pp[row * 32 + ln]      = (bf16_t)s0[r];
        pp[row * 32 + 16 + ln] = (bf16_t)s1[r];
      }
      if (ln == 0) {
        float* sc = ls_scl + qg * 16;
#pragma unroll
        for (int r = 0; r < 8; ++r) sc[r + 8 * half] = scl[r];
      }
    }
    __syncthreads();

    // ---- rescale O and accumulate O += P @ V (both waves) ----
    {
      const float* sc = ls_scl + qg * 16;
      float ms[8];
#pragma unroll
      for (int r = 0; r < 8; ++r) ms[r] = sc[r + 8 * half];
#pragma unroll
      for (int t = 0; t < 16; ++t)
#pragma unroll
        for (int r = 0; r < 8; ++r) o[t][r] *= ms[r];

      const bf16_t* pp = ls_p + qg * 512;
      v16bf pa = ld_frag(pp + ln * 32 + 16 * half, pp + ln * 32 + 16 * half + 8);
#pragma unroll
      for (int t = 0; t < 16; ++t) {
        int dglob = h * 256 + t * 16 + ln;
        v16bf vb = *(const v16bf*)(lv + dglob * 32 + 16 * half);
        o[t] = WMMA_BF16(pa, vb, o[t]);
      }
    }

    // commit next V tile into its (free) buffer, wait for async K, fence tile
    if (more) store_v_lds(ls_v + nxt * (32 * Dc), tid, vreg);
    async_wait();
    __syncthreads();
  }

  // ---- finalize: divide by softmax denominator, write out ----
  if (h == 1 && ln == 0) {
    float* sc = ls_scl + qg * 16;
#pragma unroll
    for (int r = 0; r < 8; ++r) sc[r + 8 * half] = 1.0f / lrow[r];
  }
  __syncthreads();
  {
    const float* sc = ls_scl + qg * 16;
    float inv[8];
#pragma unroll
    for (int r = 0; r < 8; ++r) inv[r] = sc[r + 8 * half];
#pragma unroll
    for (int t = 0; t < 16; ++t) {
      int col = h * 256 + t * 16 + ln;
#pragma unroll
      for (int r = 0; r < 8; ++r) {
        int row = q0 + qg * 16 + r + 8 * half;
        out[((size_t)b * Qc + row) * Dc + col] = o[t][r] * inv[r];
      }
    }
  }
}

// ---------------------------------------------------------------------------
extern "C" void kernel_launch(void* const* d_in, const int* in_sizes, int n_in,
                              void* d_out, int out_size, void* d_ws, size_t ws_size,
                              hipStream_t stream) {
  (void)in_sizes; (void)n_in; (void)out_size; (void)ws_size;
  const float* queries = (const float*)d_in[0];
  const float* keys    = (const float*)d_in[1];
  const float* values  = (const float*)d_in[2];
  const float* W       = (const float*)d_in[3];
  const int*   masks   = (const int*)d_in[4];
  float* out = (float*)d_out;

  // workspace: Kbf | Vbf | QWbf | Qbf (16MB each) | Wt (512KB)
  const size_t SZ = (size_t)Bc * Kc * Dc * sizeof(bf16_t);
  char* ws = (char*)d_ws;
  bf16_t* kbf  = (bf16_t*)(ws);
  bf16_t* vbf  = (bf16_t*)(ws + SZ);
  bf16_t* qwbf = (bf16_t*)(ws + 2 * SZ);
  bf16_t* qbf  = (bf16_t*)(ws + 3 * SZ);
  bf16_t* wtbf = (bf16_t*)(ws + 4 * SZ);

  const int n8 = Bc * Kc * Dc / 8;   // 8 elems per thread
  cvt_kernel<<<n8 / 256, 256, 0, stream>>>(keys,    kbf, n8);
  cvt_kernel<<<n8 / 256, 256, 0, stream>>>(values,  vbf, n8);
  cvt_kernel<<<n8 / 256, 256, 0, stream>>>(queries, qbf, n8);
  wt_kernel<<<Dc * Dc / 256, 256, 0, stream>>>(W, wtbf);

  qw_kernel<<<dim3(Bc * Qc / 16, 4), 256, 0, stream>>>(qbf, wtbf, qwbf);

  (void)hipFuncSetAttribute((const void*)fa_kernel,
                            hipFuncAttributeMaxDynamicSharedMemorySize,
                            (int)SM_TOT);
  fa_kernel<<<Bc * Qc / 64, 256, SM_TOT, stream>>>(qwbf, kbf, vbf, masks, out);
}
